// BloomAttention_49727131353949
// MI455X (gfx1250) — compile-verified, matrix operand order checked by reading, not run on hardware
//
#include <hip/hip_runtime.h>

// ---------------------------------------------------------------------------
// BLOOM attention block for MI455X (gfx1250), bf16 WMMA + flash attention.
// B=2, S=2048, HID=4096, H=32, HD=128.
// Pipeline:
//   1. hidden fp32 -> bf16
//   2. Wqkv, Wd fp32 -> bf16 transposed (so GEMM B-operand rows are contiguous)
//   3. QKV = X @ Wqkv + bqkv            (bf16 WMMA, f32 accum, bf16 out)
//   4. flash attention per (b, h, 128-q-row tile) with alibi + causal mask
//   5. out = residual + CTX @ Wd + bd   (bf16 WMMA, f32 accum, f32 out)
// attention_mask input (d_in[3]) is the fixed causal mask; reproduced
// analytically from indices (deterministic, avoids bool-dtype ambiguity).
// ---------------------------------------------------------------------------

typedef __attribute__((ext_vector_type(16))) __bf16 v16bf;
typedef __attribute__((ext_vector_type(8)))  float  v8f;

// 16-bit WMMA A/B fragment: element e <-> K = (e>>3)*16 + laneHi*8 + (e&7).
// Per lane this is two contiguous 16-byte chunks at kBase+laneHi*8 (halves)
// and +16 halves. Works for LDS and global pointers (addrspace inferred).
__device__ __forceinline__ v16bf load_frag(const __bf16* p) {
  union { uint4 u[2]; v16bf v; } f;
  f.u[0] = *reinterpret_cast<const uint4*>(p);
  f.u[1] = *reinterpret_cast<const uint4*>(p + 16);
  return f.v;
}

__device__ __forceinline__ v8f wmma_bf16(v16bf a, v16bf b, v8f c) {
  return __builtin_amdgcn_wmma_f32_16x16x32_bf16(false, a, false, b, (short)0,
                                                 c, false, false);
}

// reductions across the 16 lanes sharing a C/D-matrix row (wave32: lanes 0-15
// and 16-31 hold different M halves, xor masks <16 stay inside each half)
__device__ __forceinline__ float rowmax16(float v) {
#pragma unroll
  for (int m = 8; m >= 1; m >>= 1) v = fmaxf(v, __shfl_xor(v, m, 32));
  return v;
}
__device__ __forceinline__ float rowsum16(float v) {
#pragma unroll
  for (int m = 8; m >= 1; m >>= 1) v += __shfl_xor(v, m, 32);
  return v;
}

// ------------------------------ converts -----------------------------------

__global__ __launch_bounds__(256) void cvt_bf16_kernel(
    const float* __restrict__ in, __bf16* __restrict__ out, long n) {
  long i = ((long)blockIdx.x * 256 + threadIdx.x) * 4;
  if (i + 3 >= n + 3) return;
  if (i + 3 < n) {
    float4 v = *reinterpret_cast<const float4*>(in + i);
    union { __bf16 h[4]; uint2 u; } o;
    o.h[0] = (__bf16)v.x; o.h[1] = (__bf16)v.y;
    o.h[2] = (__bf16)v.z; o.h[3] = (__bf16)v.w;
    *reinterpret_cast<uint2*>(out + i) = o.u;
  }
}

// out[n][k] = (bf16) in[k][n];  in is [K x N] row-major fp32
__global__ __launch_bounds__(256) void cvt_wT_kernel(
    const float* __restrict__ in, __bf16* __restrict__ out, int K, int N) {
  __shared__ float t[32][33];
  const int bx = blockIdx.x * 32;  // n
  const int by = blockIdx.y * 32;  // k
  const int lx = threadIdx.x & 31;
  const int ly = threadIdx.x >> 5;  // 0..7
#pragma unroll
  for (int i = 0; i < 32; i += 8)
    t[ly + i][lx] = in[(long)(by + ly + i) * N + bx + lx];
  __syncthreads();
#pragma unroll
  for (int i = 0; i < 32; i += 8)
    out[(long)(bx + ly + i) * K + by + lx] = (__bf16)t[lx][ly + i];
}

// ------------------------------- GEMM --------------------------------------
// C[m][n] = A[m][:] . Bt[n][:] + bias[n]  (+ res[m][n] when F32OUT)
// A: [M x K] bf16 row-major, Bt: [N x K] bf16 row-major.
// 256 threads / 8 waves; block tile 128x128; wave tile 32x64; K-chunk 32.
template <bool F32OUT>
__global__ __launch_bounds__(256) void gemm_bf16_wmma(
    const __bf16* __restrict__ A, const __bf16* __restrict__ Bt,
    const float* __restrict__ bias, const float* __restrict__ res,
    void* __restrict__ Cout, int M, int N, int K) {
  constexpr int ROW = 40;  // halves per LDS row: 16B aligned, conflict-free
  __shared__ __align__(16) __bf16 As[128 * ROW];
  __shared__ __align__(16) __bf16 Bs[128 * ROW];

  const int tid = threadIdx.x;
  const int lane = tid & 31;
  const int w = tid >> 5;
  const int laneLo = lane & 15;
  const int laneHi = lane >> 4;
  const int wm = w & 3;   // 4 row groups of 32
  const int wn = w >> 2;  // 2 col groups of 64
  const int m0 = blockIdx.y * 128;
  const int n0 = blockIdx.x * 128;

  v8f acc[2][4];
#pragma unroll
  for (int i = 0; i < 2; ++i)
#pragma unroll
    for (int j = 0; j < 4; ++j) acc[i][j] = {};

  // staging: thread t loads 32B of A row (t>>1) and 32B of Bt row (t>>1)
  const int srow = tid >> 1;
  const int scol = (tid & 1) * 16;  // halves
  const __bf16* aG = A + (long)(m0 + srow) * K + scol;
  const __bf16* bG = Bt + (long)(n0 + srow) * K + scol;

  for (int kb = 0; kb < K; kb += 32) {
    uint4 a0 = reinterpret_cast<const uint4*>(aG + kb)[0];
    uint4 a1 = reinterpret_cast<const uint4*>(aG + kb)[1];
    uint4 b0 = reinterpret_cast<const uint4*>(bG + kb)[0];
    uint4 b1 = reinterpret_cast<const uint4*>(bG + kb)[1];
    if (kb + 32 < K) {  // global_prefetch_b8 next K-chunk
      __builtin_prefetch(aG + kb + 32, 0, 1);
      __builtin_prefetch(bG + kb + 32, 0, 1);
    }
    __syncthreads();  // previous iteration's fragment reads done
    *reinterpret_cast<uint4*>(&As[srow * ROW + scol]) = a0;
    *reinterpret_cast<uint4*>(&As[srow * ROW + scol + 8]) = a1;
    *reinterpret_cast<uint4*>(&Bs[srow * ROW + scol]) = b0;
    *reinterpret_cast<uint4*>(&Bs[srow * ROW + scol + 8]) = b1;
    __syncthreads();

    v16bf af[2], bf[4];
#pragma unroll
    for (int mt = 0; mt < 2; ++mt)
      af[mt] = load_frag(&As[(wm * 32 + mt * 16 + laneLo) * ROW + laneHi * 8]);
#pragma unroll
    for (int nt = 0; nt < 4; ++nt)
      bf[nt] = load_frag(&Bs[(wn * 64 + nt * 16 + laneLo) * ROW + laneHi * 8]);
#pragma unroll
    for (int mt = 0; mt < 2; ++mt)
#pragma unroll
      for (int nt = 0; nt < 4; ++nt)
        acc[mt][nt] = wmma_bf16(af[mt], bf[nt], acc[mt][nt]);
  }

#pragma unroll
  for (int mt = 0; mt < 2; ++mt) {
#pragma unroll
    for (int nt = 0; nt < 4; ++nt) {
      const int n = n0 + wn * 64 + nt * 16 + laneLo;
      const float bv = bias[n];
#pragma unroll
      for (int r = 0; r < 8; ++r) {
        const int m = m0 + wm * 32 + mt * 16 + r + 8 * laneHi;
        const float v = acc[mt][nt][r] + bv;
        if constexpr (F32OUT) {
          reinterpret_cast<float*>(Cout)[(long)m * N + n] =
              v + res[(long)m * N + n];
        } else {
          reinterpret_cast<__bf16*>(Cout)[(long)m * N + n] = (__bf16)v;
        }
      }
    }
  }
}

// ---------------------------- flash attention -------------------------------
// qkv layout per row s: [h][ q(128) | k(128) | v(128) ] bf16 (3*HID halves).
// Block: one (b, h, 128-q-row tile); wave w owns q rows q0+w*16 .. +15.
// KV tiles of 128; online softmax; alibi added per key column; causal mask.
__global__ __launch_bounds__(256) void bloom_attn_kernel(
    const __bf16* __restrict__ qkv, const float* __restrict__ alibi,
    __bf16* __restrict__ ctx) {
  constexpr int S_ = 2048, HID_ = 4096, H_ = 32, HD_ = 128;
  constexpr int VROW = 136;  // Vt row pad (halves): 16B aligned, 4-bank spread
  constexpr int PROW = 40;   // P chunk row pad (halves)
  __shared__ __align__(16) __bf16 Vt[HD_ * VROW];    // V^T: [d][kv]
  __shared__ __align__(16) __bf16 Pb[8 * 16 * PROW]; // per-wave P 16x32 chunk

  const int tid = threadIdx.x;
  const int lane = tid & 31;
  const int w = tid >> 5;
  const int laneLo = lane & 15;
  const int laneHi = lane >> 4;
  const int qi = blockIdx.x;  // q tile (0..15)
  const int h = blockIdx.y;
  const int b = blockIdx.z;
  const int q0 = qi * 128;
  const long rowStride = 3 * HID_;  // halves
  const __bf16* qbase = qkv + (long)b * S_ * rowStride + (long)h * 3 * HD_;
  const float* alrow = alibi + (long)(b * H_ + h) * S_;
  const float inv_norm = 0.088388347648318447f;  // 1/sqrt(128)

  // Q fragments straight from global (A layout == contiguous 16B chunks)
  v16bf qf[4];
#pragma unroll
  for (int kd = 0; kd < 4; ++kd)
    qf[kd] = load_frag(qbase + (long)(q0 + w * 16 + laneLo) * rowStride +
                       kd * 32 + laneHi * 8);

  float mrow[8], lrow[8];
  v8f Oacc[8];
#pragma unroll
  for (int r = 0; r < 8; ++r) { mrow[r] = -3.0e38f; lrow[r] = 0.f; }
#pragma unroll
  for (int td = 0; td < 8; ++td) Oacc[td] = {};

  for (int kt = 0; kt <= qi; ++kt) {
    const int kv0 = kt * 128;

    // ---- stage V^T into LDS -------------------------------------------
    __syncthreads();  // previous iteration's Vt reads done
    {
      const int r = tid >> 1;           // kv row 0..127
      const int dh = (tid & 1) * 64;    // d offset 0/64
      const __bf16* vg =
          qbase + (long)(kv0 + r) * rowStride + 2 * HD_ + dh;
      union { uint4 u[8]; __bf16 hh[64]; } vb;
      const uint4* vg4 = reinterpret_cast<const uint4*>(vg);
#pragma unroll
      for (int j = 0; j < 8; ++j) vb.u[j] = vg4[j];
#pragma unroll
      for (int j = 0; j < 64; ++j) Vt[(dh + j) * VROW + r] = vb.hh[j];
    }
    __syncthreads();

    // ---- scores = Q @ K^T (K B-fragments straight from global) --------
    v8f sc[8];
#pragma unroll
    for (int t = 0; t < 8; ++t) sc[t] = {};
#pragma unroll
    for (int t = 0; t < 8; ++t)
#pragma unroll
      for (int kd = 0; kd < 4; ++kd) {
        v16bf kf =
            load_frag(qbase + (long)(kv0 + t * 16 + laneLo) * rowStride +
                      HD_ + kd * 32 + laneHi * 8);
        sc[t] = wmma_bf16(qf[kd], kf, sc[t]);
      }

    // ---- scale + alibi + causal mask ----------------------------------
    float al[8];
#pragma unroll
    for (int t = 0; t < 8; ++t) al[t] = alrow[kv0 + t * 16 + laneLo];
    const bool diag = (kt == qi);
#pragma unroll
    for (int t = 0; t < 8; ++t)
#pragma unroll
      for (int r = 0; r < 8; ++r) {
        float v = sc[t][r] * inv_norm + al[t];
        if (diag) {
          const int q = q0 + w * 16 + r + 8 * laneHi;
          const int kv = kv0 + t * 16 + laneLo;
          if (kv > q) v = -3.0e38f;
        }
        sc[t][r] = v;
      }

    // ---- online softmax (per C-row: lanes sharing (r, laneHi)) --------
#pragma unroll
    for (int r = 0; r < 8; ++r) {
      float pm = sc[0][r];
#pragma unroll
      for (int t = 1; t < 8; ++t) pm = fmaxf(pm, sc[t][r]);
      pm = rowmax16(pm);
      const float mnew = fmaxf(mrow[r], pm);
      float s = 0.f;
#pragma unroll
      for (int t = 0; t < 8; ++t) {
        const float p = __expf(sc[t][r] - mnew);
        sc[t][r] = p;
        s += p;
      }
      s = rowsum16(s);
      const float scale = __expf(mrow[r] - mnew);
      lrow[r] = lrow[r] * scale + s;
      mrow[r] = mnew;
#pragma unroll
      for (int td = 0; td < 8; ++td) Oacc[td][r] *= scale;
    }

    // ---- O += P @ V  (stage P per 32-kv chunk through per-wave LDS) ---
    __bf16* pw = &Pb[w * 16 * PROW];
#pragma unroll
    for (int kd = 0; kd < 4; ++kd) {
#pragma unroll
      for (int tt = 0; tt < 2; ++tt) {
        const int t = kd * 2 + tt;
#pragma unroll
        for (int r = 0; r < 8; ++r)
          pw[(r + 8 * laneHi) * PROW + tt * 16 + laneLo] = (__bf16)sc[t][r];
      }
      v16bf pf = load_frag(&Pb[w * 16 * PROW + laneLo * PROW + laneHi * 8]);
#pragma unroll
      for (int td = 0; td < 8; ++td) {
        v16bf vf =
            load_frag(&Vt[(td * 16 + laneLo) * VROW + kd * 32 + laneHi * 8]);
        Oacc[td] = wmma_bf16(pf, vf, Oacc[td]);
      }
    }
  }

  // ---- normalize and write ctx[b][q][h*HD + d] bf16 --------------------
#pragma unroll
  for (int td = 0; td < 8; ++td) {
#pragma unroll
    for (int r = 0; r < 8; ++r) {
      const int q = q0 + w * 16 + r + 8 * laneHi;
      const int d = td * 16 + laneLo;
      const float o = Oacc[td][r] / lrow[r];
      ctx[((long)b * S_ + q) * HID_ + h * HD_ + d] = (__bf16)o;
    }
  }
}

// ------------------------------ launcher -----------------------------------

extern "C" void kernel_launch(void* const* d_in, const int* in_sizes, int n_in,
                              void* d_out, int out_size, void* d_ws,
                              size_t ws_size, hipStream_t stream) {
  (void)in_sizes; (void)n_in; (void)out_size; (void)ws_size;
  constexpr int B = 2, S = 2048, HID = 4096, H = 32;
  constexpr long MR = (long)B * S;  // 4096 rows

  const float* hidden   = (const float*)d_in[0];
  const float* residual = (const float*)d_in[1];
  const float* alibi    = (const float*)d_in[2];
  // d_in[3] = attention_mask (fixed causal) — reproduced analytically
  const float* Wqkv = (const float*)d_in[4];
  const float* bqkv = (const float*)d_in[5];
  const float* Wd   = (const float*)d_in[6];
  const float* bd   = (const float*)d_in[7];
  float* out = (float*)d_out;

  char* ws = (char*)d_ws;
  const size_t MB = 1024ull * 1024ull;
  __bf16* Xbf   = (__bf16*)(ws);                 //  32 MiB [4096 x 4096]
  __bf16* WqkvT = (__bf16*)(ws + 32 * MB);       //  96 MiB [12288 x 4096]
  __bf16* WdT   = (__bf16*)(ws + 128 * MB);      //  32 MiB [4096 x 4096]
  __bf16* QKV   = (__bf16*)(ws + 160 * MB);      //  96 MiB [4096 x 12288]
  __bf16* CTX   = (__bf16*)(ws + 256 * MB);      //  32 MiB [4096 x 4096]

  // 1. converts
  {
    const long n = MR * HID;  // 16M
    cvt_bf16_kernel<<<(int)(n / 4 / 256), 256, 0, stream>>>(hidden, Xbf, n);
    cvt_wT_kernel<<<dim3(3 * HID / 32, HID / 32), 256, 0, stream>>>(
        Wqkv, WqkvT, HID, 3 * HID);
    cvt_wT_kernel<<<dim3(HID / 32, HID / 32), 256, 0, stream>>>(
        Wd, WdT, HID, HID);
  }
  // 2. QKV projection: [4096 x 4096] @ [4096 x 12288] + bqkv -> bf16
  gemm_bf16_wmma<false><<<dim3(3 * HID / 128, (int)(MR / 128)), 256, 0,
                          stream>>>(Xbf, WqkvT, bqkv, nullptr, QKV, (int)MR,
                                    3 * HID, HID);
  // 3. flash attention -> CTX bf16 [4096 x 4096]
  bloom_attn_kernel<<<dim3(S / 128, H, B), 256, 0, stream>>>(QKV, alibi, CTX);
  // 4. output projection + bias + residual -> fp32 out
  gemm_bf16_wmma<true><<<dim3(HID / 128, (int)(MR / 128)), 256, 0, stream>>>(
      CTX, WdT, bd, residual, out, (int)MR, HID, HID);
}
